// LineSAGE_30442728194375
// MI455X (gfx1250) — matrix operand order repaired and checked by reading, not compiled, not run on hardware
//
#include <hip/hip_runtime.h>

#define N_NODES 50000
#define N_EDGES 800000
#define D_IN    128
#define D_HID   256
#define D_OUT   256

typedef __attribute__((ext_vector_type(16))) __bf16 v16bf;
typedef __attribute__((ext_vector_type(8)))  float  v8f;

static __device__ __forceinline__ unsigned short f2bf(float f) {
    unsigned u = __float_as_uint(f);
    unsigned r = u + 0x7FFFu + ((u >> 16) & 1u);   // round-to-nearest-even
    return (unsigned short)(r >> 16);
}
static __device__ __forceinline__ float bf2f(unsigned short h) {
    return __uint_as_float(((unsigned)h) << 16);
}

// ---------------- utility kernels ----------------

__global__ void k_zero_f32(float* __restrict__ p, int n) {
    int i = blockIdx.x * 256 + threadIdx.x;
    if (i < n) p[i] = 0.0f;
}

__global__ void k_degree(const int* __restrict__ dst, float* __restrict__ deg) {
    int e = blockIdx.x * 256 + threadIdx.x;
    if (e < N_EDGES) atomicAdd(&deg[dst[e]], 1.0f);
}

// scatter-add f32 features: 32 threads/edge, 4 floats/thread (D_IN = 128)
__global__ void k_scatter_f32(const float* __restrict__ x,
                              const int* __restrict__ src,
                              const int* __restrict__ dst,
                              float* __restrict__ sum) {
    int t = blockIdx.x * 256 + threadIdx.x;      // E*32 threads
    int e = t >> 5;
    int f = (t & 31) * 4;
    int s = src[e], d = dst[e];
    float4 v = *(const float4*)(x + (size_t)s * D_IN + f);
    float* o = sum + (size_t)d * D_IN + f;
    atomicAdd(o + 0, v.x); atomicAdd(o + 1, v.y);
    atomicAdd(o + 2, v.z); atomicAdd(o + 3, v.w);
}

// scatter-add bf16 features: 32 threads/edge, 8 bf16/thread (D_HID = 256)
__global__ void k_scatter_bf16(const unsigned short* __restrict__ h,
                               const int* __restrict__ src,
                               const int* __restrict__ dst,
                               float* __restrict__ sum) {
    int t = blockIdx.x * 256 + threadIdx.x;      // E*32 threads
    int e = t >> 5;
    int f = (t & 31) * 8;
    int s = src[e], d = dst[e];
    uint4 v = *(const uint4*)(h + (size_t)s * D_HID + f);  // 8 bf16
    float* o = sum + (size_t)d * D_HID + f;
    atomicAdd(o + 0, bf2f((unsigned short)(v.x & 0xFFFF)));
    atomicAdd(o + 1, bf2f((unsigned short)(v.x >> 16)));
    atomicAdd(o + 2, bf2f((unsigned short)(v.y & 0xFFFF)));
    atomicAdd(o + 3, bf2f((unsigned short)(v.y >> 16)));
    atomicAdd(o + 4, bf2f((unsigned short)(v.z & 0xFFFF)));
    atomicAdd(o + 5, bf2f((unsigned short)(v.z >> 16)));
    atomicAdd(o + 6, bf2f((unsigned short)(v.w & 0xFFFF)));
    atomicAdd(o + 7, bf2f((unsigned short)(v.w >> 16)));
}

__global__ void k_cvt_bf16(const float* __restrict__ in,
                           unsigned short* __restrict__ out, int n) {
    int i = blockIdx.x * 256 + threadIdx.x;
    if (i < n) out[i] = f2bf(in[i]);
}

// out[i] = bf16( sum[i] / max(deg[i/d], 1) ), d is a power of two
__global__ void k_norm_bf16(const float* __restrict__ sum,
                            const float* __restrict__ deg,
                            unsigned short* __restrict__ out,
                            int d_log2, int n) {
    int i = blockIdx.x * 256 + threadIdx.x;
    if (i >= n) return;
    float dg = deg[i >> d_log2];
    dg = dg < 1.0f ? 1.0f : dg;
    out[i] = f2bf(sum[i] / dg);
}

// W[K][Nd] (f32, row-major) -> Wt[Nd][K] (bf16)
__global__ void k_transpose_w(const float* __restrict__ W,
                              unsigned short* __restrict__ Wt, int K, int Nd) {
    int i = blockIdx.x * 256 + threadIdx.x;
    if (i >= K * Nd) return;
    int k = i / Nd, n = i % Nd;
    Wt[n * K + k] = f2bf(W[i]);
}

// ---------------- async copy: contiguous global -> LDS ----------------
// n128 must be a multiple of nthreads/ iterations exact so EXEC stays uniform.
static __device__ __forceinline__ void async_copy_b128(
    const void* g, unsigned lds_byte_off, int n128, int tid, int nthreads) {
    for (int i = tid; i < n128; i += nthreads) {
        unsigned long long ga = (unsigned long long)g + (unsigned long long)i * 16;
        unsigned la = lds_byte_off + (unsigned)i * 16u;
        asm volatile("global_load_async_to_lds_b128 %0, %1, off"
                     :: "v"(la), "v"(ga) : "memory");
    }
}
static __device__ __forceinline__ void async_wait0() {
    asm volatile("s_wait_asynccnt 0x0" ::: "memory");
}
static __device__ __forceinline__ unsigned lds_off(const void* p) {
    return (unsigned)(unsigned long long)p;   // low 32 bits = LDS byte address
}

// ---------------- WMMA GEMM core ----------------
// One wave computes a 16(M) x 64(N) strip: A fragment loaded once per K-step
// from LDS (ds_load_b128), reused across 4 WMMAs against 4 B column-tiles.
// A layout (16-bit, 16x32/step): lo lanes K={0..7,16..23}, hi lanes K={8..15,24..31}
// B layout (16-bit, 32x16/step): lo lanes K={0..7,16..23}, hi lanes K={8..15,24..31}
template <int K>
static __device__ __forceinline__ void wmma_strip(
    const unsigned short* __restrict__ sA,   // LDS tile [16][K]
    const unsigned short* __restrict__ Bt,   // global [256][K] (bf16, N-major)
    int n0, int l16, int hi, v8f acc[4])
{
    const unsigned short* arow = sA + l16 * K;
#pragma unroll
    for (int kk = 0; kk < K; kk += 32) {
        union { v16bf v; uint4 q[2]; } ua;
        const int kbA = kk + hi * 8;
        ua.q[0] = *(const uint4*)(arow + kbA);        // K pairs 0..7 of this half
        ua.q[1] = *(const uint4*)(arow + kbA + 16);   // K pairs 16..23 of this half
        const int kbB = kk + hi * 8;
#pragma unroll
        for (int t = 0; t < 4; ++t) {
            union { v16bf v; uint4 q[2]; } ub;
            const unsigned short* brow = Bt + (size_t)(n0 + 16 * t + l16) * K + kbB;
            ub.q[0] = *(const uint4*)(brow);
            ub.q[1] = *(const uint4*)(brow + 16);
            acc[t] = __builtin_amdgcn_wmma_f32_16x16x32_bf16(
                         false, ua.v, false, ub.v, (short)0, acc[t], false, false);
        }
    }
}

// Layer 1: h = relu(x@Ws1 + agg1@Wn1 + b1), bf16 out. 128 thr (4 waves), grid 3125.
__global__ void __launch_bounds__(128) k_gemm1(
        const unsigned short* __restrict__ xb,
        const unsigned short* __restrict__ aggb,
        const unsigned short* __restrict__ Wst,
        const unsigned short* __restrict__ Wnt,
        const float* __restrict__ b1,
        unsigned short* __restrict__ hout) {
    __shared__ __align__(16) unsigned short sX[16 * D_IN];
    __shared__ __align__(16) unsigned short sG[16 * D_IN];
    const int tid = threadIdx.x;
    const int m0 = blockIdx.x * 16;
    // 16 rows are contiguous: flat 4KB copies into LDS via async DMA
    async_copy_b128(xb   + (size_t)m0 * D_IN, lds_off(sX), (16 * D_IN * 2) / 16, tid, 128);
    async_copy_b128(aggb + (size_t)m0 * D_IN, lds_off(sG), (16 * D_IN * 2) / 16, tid, 128);
    async_wait0();
    __syncthreads();

    const int lane = tid & 31, l16 = lane & 15, hi = lane >> 4;
    const int n0 = (tid >> 5) * 64;
    v8f acc[4] = {v8f{}, v8f{}, v8f{}, v8f{}};
    wmma_strip<D_IN>(sX, Wst, n0, l16, hi, acc);
    wmma_strip<D_IN>(sG, Wnt, n0, l16, hi, acc);

    const int mbase = m0 + (hi << 3);
#pragma unroll
    for (int t = 0; t < 4; ++t) {
        const int n = n0 + 16 * t + l16;
        const float bias = b1[n];
#pragma unroll
        for (int r = 0; r < 8; ++r) {
            float v = acc[t][r] + bias;
            v = v > 0.0f ? v : 0.0f;
            hout[(size_t)(mbase + r) * D_HID + n] = f2bf(v);
        }
    }
}

// Layer 2: h2 = x@Wres + h@Ws2 + agg2@Wn2 + b2, f32 out. 128 thr, grid 3125.
__global__ void __launch_bounds__(128) k_gemm2(
        const unsigned short* __restrict__ xb,
        const unsigned short* __restrict__ hb,
        const unsigned short* __restrict__ agg2b,
        const unsigned short* __restrict__ Wrt,
        const unsigned short* __restrict__ Ws2t,
        const unsigned short* __restrict__ Wn2t,
        const float* __restrict__ b2,
        float* __restrict__ h2) {
    __shared__ __align__(16) unsigned short sX[16 * D_IN];
    __shared__ __align__(16) unsigned short sH[16 * D_HID];
    __shared__ __align__(16) unsigned short sG[16 * D_HID];
    const int tid = threadIdx.x;
    const int m0 = blockIdx.x * 16;
    async_copy_b128(xb    + (size_t)m0 * D_IN,  lds_off(sX), (16 * D_IN  * 2) / 16, tid, 128);
    async_copy_b128(hb    + (size_t)m0 * D_HID, lds_off(sH), (16 * D_HID * 2) / 16, tid, 128);
    async_copy_b128(agg2b + (size_t)m0 * D_HID, lds_off(sG), (16 * D_HID * 2) / 16, tid, 128);
    async_wait0();
    __syncthreads();

    const int lane = tid & 31, l16 = lane & 15, hi = lane >> 4;
    const int n0 = (tid >> 5) * 64;
    v8f acc[4] = {v8f{}, v8f{}, v8f{}, v8f{}};
    wmma_strip<D_IN >(sX, Wrt,  n0, l16, hi, acc);
    wmma_strip<D_HID>(sH, Ws2t, n0, l16, hi, acc);
    wmma_strip<D_HID>(sG, Wn2t, n0, l16, hi, acc);

    const int mbase = m0 + (hi << 3);
#pragma unroll
    for (int t = 0; t < 4; ++t) {
        const int n = n0 + 16 * t + l16;
        const float bias = b2[n];
#pragma unroll
        for (int r = 0; r < 8; ++r)
            h2[(size_t)(mbase + r) * D_OUT + n] = acc[t][r] + bias;
    }
}

// MLP head: out[n,0:2] = h2[n,:] @ W_mlp + b_mlp (N=2 too narrow for WMMA)
__global__ void k_head(const float* __restrict__ h2,
                       const float* __restrict__ Wm,
                       const float* __restrict__ bm,
                       float* __restrict__ out) {
    int n = blockIdx.x * 256 + threadIdx.x;
    if (n >= N_NODES) return;
    const float* row = h2 + (size_t)n * D_OUT;
    float a0 = bm[0], a1 = bm[1];
#pragma unroll 4
    for (int k = 0; k < D_OUT; ++k) {
        float v = row[k];
        a0 += v * Wm[k * 2 + 0];
        a1 += v * Wm[k * 2 + 1];
    }
    out[n * 2 + 0] = a0;
    out[n * 2 + 1] = a1;
}

// ---------------- launch ----------------

extern "C" void kernel_launch(void* const* d_in, const int* in_sizes, int n_in,
                              void* d_out, int out_size, void* d_ws, size_t ws_size,
                              hipStream_t stream) {
    (void)in_sizes; (void)n_in; (void)out_size; (void)ws_size;
    const float* x       = (const float*)d_in[0];
    const int*   src     = (const int*)  d_in[1];
    const int*   dst     = (const int*)  d_in[2];
    const float* W_self1 = (const float*)d_in[3];
    const float* W_neigh1= (const float*)d_in[4];
    const float* b1      = (const float*)d_in[5];
    const float* W_self2 = (const float*)d_in[6];
    const float* W_neigh2= (const float*)d_in[7];
    const float* b2      = (const float*)d_in[8];
    const float* W_res   = (const float*)d_in[9];
    const float* W_mlp   = (const float*)d_in[10];
    const float* b_mlp   = (const float*)d_in[11];
    float* out = (float*)d_out;

    // workspace carve-up (256B aligned)
    char* ws = (char*)d_ws;
    size_t off = 0;
    auto carve = [&](size_t bytes) -> char* {
        char* p = ws + off;
        off += (bytes + 255) & ~(size_t)255;
        return p;
    };
    float*          deg    = (float*)         carve((size_t)N_NODES * 4);
    float*          sum1   = (float*)         carve((size_t)N_NODES * D_IN  * 4);
    float*          sum2   = (float*)         carve((size_t)N_NODES * D_HID * 4); // reused as h2
    unsigned short* x_bf   = (unsigned short*)carve((size_t)N_NODES * D_IN  * 2);
    unsigned short* agg1_bf= (unsigned short*)carve((size_t)N_NODES * D_IN  * 2);
    unsigned short* h_bf   = (unsigned short*)carve((size_t)N_NODES * D_HID * 2);
    unsigned short* agg2_bf= (unsigned short*)carve((size_t)N_NODES * D_HID * 2);
    unsigned short* Ws1t   = (unsigned short*)carve((size_t)D_IN  * D_HID * 2);
    unsigned short* Wn1t   = (unsigned short*)carve((size_t)D_IN  * D_HID * 2);
    unsigned short* Wrt    = (unsigned short*)carve((size_t)D_IN  * D_OUT * 2);
    unsigned short* Ws2t   = (unsigned short*)carve((size_t)D_HID * D_OUT * 2);
    unsigned short* Wn2t   = (unsigned short*)carve((size_t)D_HID * D_OUT * 2);
    float* h2 = sum2;  // sum2 free after agg2_bf extraction; reuse for h2

    const int T = 256;
    const int nIn  = N_NODES * D_IN;   // 6.4M
    const int nHid = N_NODES * D_HID;  // 12.8M
    const int eThreads = N_EDGES * 32; // 25.6M

    // 1) degree + layer-1 aggregation (f32 atomics, L2-resident)
    k_zero_f32<<<(N_NODES + T - 1) / T, T, 0, stream>>>(deg, N_NODES);
    k_zero_f32<<<nIn / T, T, 0, stream>>>(sum1, nIn);
    k_degree<<<(N_EDGES + T - 1) / T, T, 0, stream>>>(dst, deg);
    k_scatter_f32<<<eThreads / T, T, 0, stream>>>(x, src, dst, sum1);

    // 2) bf16 conversions + weight transposes
    k_cvt_bf16<<<nIn / T, T, 0, stream>>>(x, x_bf, nIn);
    k_norm_bf16<<<nIn / T, T, 0, stream>>>(sum1, deg, agg1_bf, 7, nIn);
    k_transpose_w<<<(D_IN * D_HID) / T, T, 0, stream>>>(W_self1,  Ws1t, D_IN,  D_HID);
    k_transpose_w<<<(D_IN * D_HID) / T, T, 0, stream>>>(W_neigh1, Wn1t, D_IN,  D_HID);
    k_transpose_w<<<(D_IN * D_OUT) / T, T, 0, stream>>>(W_res,    Wrt,  D_IN,  D_OUT);
    k_transpose_w<<<(D_HID * D_OUT) / T, T, 0, stream>>>(W_self2,  Ws2t, D_HID, D_OUT);
    k_transpose_w<<<(D_HID * D_OUT) / T, T, 0, stream>>>(W_neigh2, Wn2t, D_HID, D_OUT);

    // 3) layer-1 WMMA GEMM (async A tile -> LDS, dual product + bias + relu)
    k_gemm1<<<N_NODES / 16, 128, 0, stream>>>(x_bf, agg1_bf, Ws1t, Wn1t, b1, h_bf);

    // 4) layer-2 aggregation
    k_zero_f32<<<nHid / T, T, 0, stream>>>(sum2, nHid);
    k_scatter_bf16<<<eThreads / T, T, 0, stream>>>(h_bf, src, dst, sum2);
    k_norm_bf16<<<nHid / T, T, 0, stream>>>(sum2, deg, agg2_bf, 8, nHid);

    // 5) layer-2 WMMA GEMM (async A tiles -> LDS, triple product + bias)
    k_gemm2<<<N_NODES / 16, 128, 0, stream>>>(
        x_bf, h_bf, agg2_bf, Wrt, Ws2t, Wn2t, b2, h2);

    // 6) MLP head
    k_head<<<(N_NODES + T - 1) / T, T, 0, stream>>>(h2, W_mlp, b_mlp, out);
}
